// GraphConvolutionLayer_47261820125198
// MI455X (gfx1250) — matrix operand order repaired and checked by reading, not compiled, not run on hardware
//
#include <hip/hip_runtime.h>

#define N    8192
#define DIN  128
#define DOUT 64
#define KSTEP 32
#define BLK_ROWS 64   // 4 waves * 16 rows

typedef __attribute__((ext_vector_type(16))) __bf16 v16bf;
typedef __attribute__((ext_vector_type(8)))  float  v8f;

// ---------------- Kernel 1: deg = rowsum(A); dinv = rsqrt(deg) ----------------
__global__ __launch_bounds__(256) void k_rowsum_dinv(const float* __restrict__ A,
                                                     float* __restrict__ dinv) {
    const int wave = threadIdx.x >> 5;
    const int lane = threadIdx.x & 31;
    const int row  = blockIdx.x * 8 + wave;   // 1024 blocks * 8 rows
    const float4* p = (const float4*)(A + (size_t)row * N);
    float s = 0.f;
    #pragma unroll 4
    for (int i = 0; i < N / 4; i += 32) {     // 64 iterations, contiguous per wave
        float4 v = p[i + lane];
        s += v.x + v.y + v.z + v.w;
    }
    #pragma unroll
    for (int off = 16; off > 0; off >>= 1) s += __shfl_down(s, off, 32);
    if (lane == 0) dinv[row] = rsqrtf(s);
}

// ------- Kernel 2: XsT[col][j] = split_bf16( X[j][col] * dinv[j] ) (transposed) -------
__global__ __launch_bounds__(256) void k_build_xst(const float* __restrict__ X,
                                                   const float* __restrict__ dinv,
                                                   unsigned short* __restrict__ xh_,
                                                   unsigned short* __restrict__ xl_) {
    __bf16* xh = (__bf16*)xh_;
    __bf16* xl = (__bf16*)xl_;
    const int id  = blockIdx.x * blockDim.x + threadIdx.x;  // 128*8192 threads
    const int col = id >> 13;          // /8192
    const int j   = id & (N - 1);
    float v = X[(size_t)j * DIN + col] * dinv[j];
    __bf16 h = (__bf16)v;
    __bf16 l = (__bf16)(v - (float)h);
    xh[(size_t)col * N + j] = h;
    xl[(size_t)col * N + j] = l;
}

// ---------------- Kernel 3: H = A @ XsT^T (bf16x3 WMMA), out = relu(dinv*H @ W) ----------------
__global__ __launch_bounds__(128) void k_gcn_main(const float* __restrict__ A,
                                                  const unsigned short* __restrict__ xh_,
                                                  const unsigned short* __restrict__ xl_,
                                                  const float* __restrict__ dinv,
                                                  const float* __restrict__ W,
                                                  float* __restrict__ out) {
    // stage: [buf][hi/lo][col][kk] bf16 -> 32 KB ; Hs: 64x128 f32 -> 32 KB
    __shared__ __bf16 stage[2][2][DIN][KSTEP];
    __shared__ float  Hs[BLK_ROWS][DIN];

    const int tid  = threadIdx.x;
    const int wave = tid >> 5;
    const int lane = tid & 31;
    const int m    = lane & 15;        // row within 16-row tile / col within 16-col tile
    const int hi16 = lane >> 4;        // upper half-wave selector
    const int rowbase = blockIdx.x * BLK_ROWS + wave * 16;

    // A operand addressing (contiguous k order, same as B staging):
    // lane<16: row m, K = {0..15}; lane>=16: row m, K = {16..31}
    const float* arow = A + (size_t)(rowbase + m) * N + hi16 * 16;

    v8f zero = {};
    v8f acc[8];
    #pragma unroll
    for (int t = 0; t < 8; ++t) acc[t] = zero;

    // B staging: thread tid owns column tid (128 threads == 128 cols)
    const __bf16* gh = (const __bf16*)xh_ + (size_t)tid * N;
    const __bf16* gl = (const __bf16*)xl_ + (size_t)tid * N;

    { // preload buffer 0
        const uint4* sh = (const uint4*)gh;
        const uint4* sl = (const uint4*)gl;
        uint4* dh = (uint4*)&stage[0][0][tid][0];
        uint4* dl = (uint4*)&stage[0][1][tid][0];
        #pragma unroll
        for (int q = 0; q < 4; ++q) { dh[q] = sh[q]; dl[q] = sl[q]; }
    }
    __syncthreads();

    const int NSTEP = N / KSTEP;   // 256
    for (int s = 0; s < NSTEP; ++s) {
        const int kbase = s * KSTEP;

        // prefetch next B chunk into registers (overlaps WMMA below)
        uint4 ph[4], pl[4];
        if (s + 1 < NSTEP) {
            const uint4* sh = (const uint4*)(gh + kbase + KSTEP);
            const uint4* sl = (const uint4*)(gl + kbase + KSTEP);
            #pragma unroll
            for (int q = 0; q < 4; ++q) { ph[q] = sh[q]; pl[q] = sl[q]; }
        }

        // load A fragment: 16 fp32 per lane, split into bf16 hi/lo
        float a[16];
        const float4* g0 = (const float4*)(arow + kbase);        // K group 0
        const float4* g1 = (const float4*)(arow + kbase + 8);    // K group 1
        *(float4*)&a[0]  = g0[0];
        *(float4*)&a[4]  = g0[1];
        *(float4*)&a[8]  = g1[0];
        *(float4*)&a[12] = g1[1];
        v16bf ah, al;
        #pragma unroll
        for (int e = 0; e < 16; ++e) {
            __bf16 h = (__bf16)a[e];
            ah[e] = h;
            al[e] = (__bf16)(a[e] - (float)h);
        }

        const int buf = s & 1;
        #pragma unroll
        for (int t = 0; t < 8; ++t) {
            const int col = t * 16 + m;
            // B operand (32x16 bf16): lane<16 holds K 0..15, lane>=16 holds K 16..31
            v16bf bh = *(const v16bf*)&stage[buf][0][col][hi16 * 16];
            v16bf bl = *(const v16bf*)&stage[buf][1][col][hi16 * 16];
            acc[t] = __builtin_amdgcn_wmma_f32_16x16x32_bf16(false, ah, false, bh,
                                                             (short)0, acc[t], false, false);
            acc[t] = __builtin_amdgcn_wmma_f32_16x16x32_bf16(false, ah, false, bl,
                                                             (short)0, acc[t], false, false);
            acc[t] = __builtin_amdgcn_wmma_f32_16x16x32_bf16(false, al, false, bh,
                                                             (short)0, acc[t], false, false);
        }

        __syncthreads();
        if (s + 1 < NSTEP) {
            uint4* dh = (uint4*)&stage[(s + 1) & 1][0][tid][0];
            uint4* dl = (uint4*)&stage[(s + 1) & 1][1][tid][0];
            #pragma unroll
            for (int q = 0; q < 4; ++q) { dh[q] = ph[q]; dl[q] = pl[q]; }
        }
        __syncthreads();
    }

    // ---- epilogue ----
    // stage buffer is dead now; reuse it for W (128x64 f32 == 32 KB exactly)
    float* Ws = (float*)&stage[0][0][0][0];
    for (int i = tid; i < DIN * DOUT; i += 128) Ws[i] = W[i];

    // C layout: vgpr r -> M = r + 8*hi16 ; N = t*16 + m. Scale rows by dinv, dump to LDS.
    #pragma unroll
    for (int r = 0; r < 8; ++r) {
        const int mloc = r + hi16 * 8;
        const float sc = dinv[rowbase + mloc];
        #pragma unroll
        for (int t = 0; t < 8; ++t) {
            Hs[wave * 16 + mloc][t * 16 + m] = acc[t][r] * sc;
        }
    }
    __syncthreads();

    // out tile 64 rows x 64 cols; thread -> col c (coalesced), 32 rows each
    const int c  = tid & 63;
    const int rh = tid >> 6;
    for (int rr = 0; rr < 32; ++rr) {
        const int row = rh * 32 + rr;
        const float* hrow = &Hs[row][0];
        float accv = 0.f;
        #pragma unroll 8
        for (int k = 0; k < DIN; ++k) accv = fmaf(hrow[k], Ws[k * DOUT + c], accv);
        out[(size_t)(blockIdx.x * BLK_ROWS + row) * DOUT + c] = fmaxf(accv, 0.f);
    }
}

extern "C" void kernel_launch(void* const* d_in, const int* in_sizes, int n_in,
                              void* d_out, int out_size, void* d_ws, size_t ws_size,
                              hipStream_t stream) {
    const float* X = (const float*)d_in[0];   // [8192,128]
    const float* A = (const float*)d_in[1];   // [8192,8192]
    const float* W = (const float*)d_in[2];   // [128,64]
    float* out = (float*)d_out;               // [8192,64]

    char* ws = (char*)d_ws;
    float* dinv = (float*)ws;                                   // 32 KB
    unsigned short* xh = (unsigned short*)(ws + 32 * 1024);     // 2 MB
    unsigned short* xl = (unsigned short*)(ws + 32 * 1024 + (size_t)N * DIN * 2); // 2 MB

    k_rowsum_dinv<<<N / 8, 256, 0, stream>>>(A, dinv);
    k_build_xst<<<(N * DIN) / 256, 256, 0, stream>>>(X, dinv, xh, xl);
    k_gcn_main<<<N / BLK_ROWS, 128, 0, stream>>>(A, xh, xl, dinv, W, out);
}
